// MixtureModelDecoder_824633721134
// MI455X (gfx1250) — matrix operand order repaired and checked by reading, not compiled, not run on hardware
//
#include <hip/hip_runtime.h>
#include <math.h>

#define NUM_MOLS 1024
#define N_ATOMS  32
#define ZD       64
#define HID      256
#define NT       10
#define NB       5
#define NEDGE    496                       // C(32,2)
#define TOTAL_ATOMS (NUM_MOLS * N_ATOMS)
#define EDGE_OUT_OFF (TOTAL_ATOMS * NT)    // atom_types flat first, then edge_types

typedef __attribute__((ext_vector_type(16))) __bf16 v16bf;
typedef __attribute__((ext_vector_type(8)))  float  v8f;

__device__ __forceinline__ __bf16 f2bf(float x) { return (__bf16)x; }

// ---- WMMA fragment loaders (v_wmma_f32_16x16x32_bf16 operands, contiguous k = 16*(lane/16) + i) ----

// A fragment: 16x32 bf16 tile from row-major LDS matrix (stride in elements).
// lane 0-15: row M=lane, K = 0..15; lane 16-31: same rows, K = 16..31.
// One contiguous 16-element run per lane -> lowers to 2x ds_load_b128.
__device__ __forceinline__ v16bf load_A(const __bf16* lds, int row_base, int stride,
                                        int k_base, int lane) {
  const int m  = lane & 15;
  const int kh = (lane >> 4) << 4;
  const __bf16* p = lds + (row_base + m) * stride + k_base + kh;
  v16bf a;
#pragma unroll
  for (int i = 0; i < 16; ++i) {
    int k = i;                         // 0..15
    a[i] = p[k];
  }
  return a;
}

// B fragment from pre-swizzled fragment-major storage:
// frag[seg][lane][i] with lane = (n&15)|(k&16), i = k&15.
// 16 contiguous bf16 per lane -> 2x ds_load_b128 (no strided gather).
__device__ __forceinline__ v16bf load_Bf(const __bf16* frag, int seg, int lane) {
  const __bf16* p = frag + (seg * 32 + lane) * 16;
  v16bf b;
#pragma unroll
  for (int i = 0; i < 16; ++i) b[i] = p[i];
  return b;
}

// B fragment where B = Z^T, Z row-major [n][k]: per-lane data is contiguous in k.
__device__ __forceinline__ v16bf load_B_T(const __bf16* ldsZ, int col_base, int zstride,
                                          int k_base, int lane) {
  const int n  = lane & 15;
  const int kh = (lane >> 4) << 4;
  const __bf16* p = ldsZ + (col_base + n) * zstride + k_base + kh;
  v16bf b;
#pragma unroll
  for (int i = 0; i < 16; ++i) b[i] = p[i];
  return b;
}

__device__ __forceinline__ v8f wmma_bf16(v16bf a, v16bf b, v8f c) {
  return __builtin_amdgcn_wmma_f32_16x16x32_bf16(false, a, false, b, (short)0, c,
                                                 false, false);
}

// -------------------- kernels --------------------

__global__ void k_init(float* ws) {
  ws[threadIdx.x] = 0.f;                 // 512 threads: sum[256] + sumsq[256]
}

__global__ __launch_bounds__(256)
void k_stats(const float* __restrict__ z, const float* __restrict__ W1,
             const float* __restrict__ b1, float* __restrict__ ws) {
  __shared__ __bf16 sW1f[ZD * HID];      // 32 KB, fragment-major: 32 segs x 32 x 16
  __shared__ __bf16 sZ[N_ATOMS * ZD];    // 4 KB, row-major
  const int tid = threadIdx.x, blk = blockIdx.x;
  // W1 row-major [k][n] (64x256) -> fragment layout, seg = (n/16)*2 + k/32
  for (int idx = tid; idx < ZD * HID; idx += 256) {
    const int k = idx >> 8, n = idx & 255;
    const int seg = ((n >> 4) << 1) | (k >> 5);
    sW1f[(seg * 32 + ((n & 15) | (k & 16))) * 16 + (k & 15)] = f2bf(W1[idx]);
  }
  const float* zb = z + (size_t)blk * N_ATOMS * ZD;
  for (int i = tid; i < N_ATOMS * ZD; i += 256) sZ[i] = f2bf(zb[i]);
  __syncthreads();

  const int lane = tid & 31, wave = tid >> 5;
  float* gsum = ws;
  float* gssq = ws + HID;

#pragma unroll
  for (int nt = 0; nt < 2; ++nt) {
    const int tile = wave * 2 + nt;              // 16-col tile index
    v8f c0 = {}, c1 = {};
#pragma unroll
    for (int ks = 0; ks < 2; ++ks) {             // K = 64 = 2 x 32
      v16bf b  = load_Bf(sW1f, tile * 2 + ks, lane);
      v16bf a0 = load_A(sZ, 0,  ZD, ks * 32, lane);
      v16bf a1 = load_A(sZ, 16, ZD, ks * 32, lane);
      c0 = wmma_bf16(a0, b, c0);
      c1 = wmma_bf16(a1, b, c1);
    }
    const int ncol = tile * 16 + (lane & 15);
    const float bv = b1[ncol];
    float ps = 0.f, pss = 0.f;
#pragma unroll
    for (int v = 0; v < 8; ++v) {
      float h0 = c0[v] + bv, h1 = c1[v] + bv;
      ps  += h0 + h1;
      pss += h0 * h0 + h1 * h1;
    }
    ps  += __shfl_xor(ps, 16, 32);               // combine row-halves (lane ^ 16)
    pss += __shfl_xor(pss, 16, 32);
    if (lane < 16) {
      atomicAdd(&gsum[ncol], ps);
      atomicAdd(&gssq[ncol], pss);
    }
  }
}

__global__ void k_bnfinal(const float* __restrict__ gamma,
                          const float* __restrict__ beta, float* __restrict__ ws) {
  const int i = threadIdx.x;                     // 256 threads
  const float invN = 1.0f / (float)TOTAL_ATOMS;
  const float mean = ws[i] * invN;
  const float var  = ws[HID + i] * invN - mean * mean;
  const float rstd = rsqrtf(var + 1e-5f);
  const float sc   = rstd * gamma[i];
  ws[2 * HID + i] = sc;
  ws[3 * HID + i] = beta[i] - mean * sc;
}

__global__ __launch_bounds__(256)
void k_atoms(const float* __restrict__ z, const float* __restrict__ W1,
             const float* __restrict__ b1, const float* __restrict__ W2,
             const float* __restrict__ b2, const float* __restrict__ ws,
             float* __restrict__ out) {
  __shared__ __bf16 sW1f[ZD * HID];      // 32 KB fragment-major
  __shared__ __bf16 sZ[N_ATOMS * ZD];    // 4 KB
  __shared__ __bf16 sG[N_ATOMS * HID];   // 16 KB gelu output, row-major (A of GEMM2)
  __shared__ __bf16 sW2f[HID * 16];      // 8 KB fragment-major, W2 padded 10->16
  const int tid = threadIdx.x, blk = blockIdx.x;
  for (int idx = tid; idx < ZD * HID; idx += 256) {
    const int k = idx >> 8, n = idx & 255;
    const int seg = ((n >> 4) << 1) | (k >> 5);
    sW1f[(seg * 32 + ((n & 15) | (k & 16))) * 16 + (k & 15)] = f2bf(W1[idx]);
  }
  const float* zb = z + (size_t)blk * N_ATOMS * ZD;
  for (int i = tid; i < N_ATOMS * ZD; i += 256) sZ[i] = f2bf(zb[i]);
  // W2 (256x10) -> fragment layout (single col tile, 8 k-steps)
  for (int idx = tid; idx < HID * 16; idx += 256) {
    const int k = idx >> 4, n = idx & 15;
    const float v = (n < NT) ? W2[k * NT + n] : 0.f;
    sW2f[(((k >> 5) * 32) + (n | (k & 16))) * 16 + (k & 15)] = f2bf(v);
  }
  __syncthreads();

  const int lane = tid & 31, wave = tid >> 5;
#pragma unroll
  for (int nt = 0; nt < 2; ++nt) {
    const int tile = wave * 2 + nt;
    v8f c0 = {}, c1 = {};
#pragma unroll
    for (int ks = 0; ks < 2; ++ks) {
      v16bf b  = load_Bf(sW1f, tile * 2 + ks, lane);
      v16bf a0 = load_A(sZ, 0,  ZD, ks * 32, lane);
      v16bf a1 = load_A(sZ, 16, ZD, ks * 32, lane);
      c0 = wmma_bf16(a0, b, c0);
      c1 = wmma_bf16(a1, b, c1);
    }
    const int ncol = tile * 16 + (lane & 15);
    const float bv = b1[ncol];
    const float sc = ws[2 * HID + ncol];
    const float sh = ws[3 * HID + ncol];
    const int mrow = (lane >> 4) << 3;
#pragma unroll
    for (int v = 0; v < 8; ++v) {
      float h0 = (c0[v] + bv) * sc + sh;
      float h1 = (c1[v] + bv) * sc + sh;
      h0 = 0.5f * h0 * (1.0f + erff(h0 * 0.70710678118f));   // exact gelu
      h1 = 0.5f * h1 * (1.0f + erff(h1 * 0.70710678118f));
      sG[(mrow + v) * HID + ncol]      = f2bf(h0);
      sG[(16 + mrow + v) * HID + ncol] = f2bf(h1);
    }
  }
  __syncthreads();

  if (wave < 2) {                                 // atom_types = g @ W2pad + b2
    const int mt = wave;
    v8f acc = {};
#pragma unroll
    for (int ks = 0; ks < 8; ++ks) {              // K = 256
      v16bf a = load_A(sG, mt * 16, HID, ks * 32, lane);
      v16bf b = load_Bf(sW2f, ks, lane);
      acc = wmma_bf16(a, b, acc);
    }
    const int n = lane & 15;
    if (n < NT) {
      const float bv = b2[n];
      const int mrow = mt * 16 + ((lane >> 4) << 3);
#pragma unroll
      for (int v = 0; v < 8; ++v)
        out[(size_t)(blk * N_ATOMS + mrow + v) * NT + n] = acc[v] + bv;
    }
  }
}

__global__ __launch_bounds__(256)
void k_edges(const float* __restrict__ z, const float* __restrict__ bond,
             float* __restrict__ out) {
  __shared__ __bf16 sWf[NB * ZD * ZD];           // 40 KB symmetrized, fragment-major
  __shared__ __bf16 sZ[N_ATOMS * ZD];            // 4 KB row-major
  __shared__ __bf16 sY[N_ATOMS * ZD];            // 4 KB row-major
  __shared__ float  sGm[NB * N_ATOMS * N_ATOMS]; // 20 KB logits
  const int tid = threadIdx.x, mol = blockIdx.x;
  const float* zb = z + (size_t)mol * N_ATOMS * ZD;
  for (int i = tid; i < N_ATOMS * ZD; i += 256) sZ[i] = f2bf(zb[i]);
  // symmetrize + swizzle: 8 segs per bond type: seg = t*8 + (c/16)*2 + r/32
  for (int idx = tid; idx < NB * ZD * ZD; idx += 256) {
    const int t = idx >> 12, r = (idx >> 6) & 63, c = idx & 63;   // r=k, c=n
    const float v = 0.5f * (bond[(t << 12) + (r << 6) + c] +
                            bond[(t << 12) + (c << 6) + r]);
    const int seg = (t << 3) | ((c >> 4) << 1) | (r >> 5);
    sWf[(seg * 32 + ((c & 15) | (r & 16))) * 16 + (r & 15)] = f2bf(v);
  }
  __syncthreads();

  const int lane = tid & 31, wave = tid >> 5;
  for (int t = 0; t < NB; ++t) {
    // GEMM1: Y = Z @ W_t (32x64 @ 64x64): 2x4 tiles, one per wave
    {
      const int mt = wave >> 2, nt = wave & 3;
      const int rowb = mt * 16;
      v8f acc = {};
#pragma unroll
      for (int ks = 0; ks < 2; ++ks) {
        v16bf a = load_A(sZ, rowb, ZD, ks * 32, lane);
        v16bf b = load_Bf(sWf, (t << 3) | (nt << 1) | ks, lane);
        acc = wmma_bf16(a, b, acc);
      }
      const int n = nt * 16 + (lane & 15);
      const int mrow = rowb + ((lane >> 4) << 3);
#pragma unroll
      for (int v = 0; v < 8; ++v) sY[(mrow + v) * ZD + n] = f2bf(acc[v]);
    }
    __syncthreads();
    // GEMM2: G_t = Y @ Z^T (32x64 @ 64x32): 2x2 tiles, waves 0..3
    if (wave < 4) {
      const int mt = wave >> 1, nt = wave & 1;
      const int col = nt * 16, rowb = mt * 16;
      v8f acc = {};
#pragma unroll
      for (int ks = 0; ks < 2; ++ks) {
        v16bf a = load_A(sY, rowb, ZD, ks * 32, lane);
        v16bf b = load_B_T(sZ, col, ZD, ks * 32, lane);
        acc = wmma_bf16(a, b, acc);
      }
      const int n = col + (lane & 15);
      const int mrow = rowb + ((lane >> 4) << 3);
      float* g = sGm + t * N_ATOMS * N_ATOMS;
#pragma unroll
      for (int v = 0; v < 8; ++v) g[(mrow + v) * N_ATOMS + n] = acc[v];
    }
    __syncthreads();                             // sY reuse + sGm visibility
  }

  // per-edge 5-way softmax; edge order = itertools.combinations (i<j, lex)
  float* eout = out + EDGE_OUT_OFF + (size_t)mol * NEDGE * NB;
  for (int e = tid; e < NEDGE; e += 256) {
    int i = 0, rem = e;
    while (rem >= (N_ATOMS - 1 - i)) { rem -= (N_ATOMS - 1 - i); ++i; }
    const int j = i + 1 + rem;
    float l[NB], mx = -1e30f;
#pragma unroll
    for (int t = 0; t < NB; ++t) {
      l[t] = sGm[t * N_ATOMS * N_ATOMS + i * N_ATOMS + j];
      mx = fmaxf(mx, l[t]);
    }
    float s = 0.f;
#pragma unroll
    for (int t = 0; t < NB; ++t) { l[t] = __expf(l[t] - mx); s += l[t]; }
    const float inv = 1.0f / s;
#pragma unroll
    for (int t = 0; t < NB; ++t) eout[e * NB + t] = l[t] * inv;
  }
}

extern "C" void kernel_launch(void* const* d_in, const int* in_sizes, int n_in,
                              void* d_out, int out_size, void* d_ws, size_t ws_size,
                              hipStream_t stream) {
  (void)in_sizes; (void)n_in; (void)out_size; (void)ws_size;
  const float* z   = (const float*)d_in[0];
  const float* W1  = (const float*)d_in[1];
  const float* b1  = (const float*)d_in[2];
  const float* gam = (const float*)d_in[3];
  const float* bet = (const float*)d_in[4];
  const float* W2  = (const float*)d_in[5];
  const float* b2  = (const float*)d_in[6];
  const float* bm  = (const float*)d_in[7];
  // d_in[8] = edge_index: not needed, edges derived analytically per molecule
  float* out = (float*)d_out;
  float* ws  = (float*)d_ws;   // [sum(256) | sumsq(256) | scale(256) | shift(256)]

  k_init   <<<1, 512, 0, stream>>>(ws);
  k_stats  <<<NUM_MOLS, 256, 0, stream>>>(z, W1, b1, ws);
  k_bnfinal<<<1, HID, 0, stream>>>(gam, bet, ws);
  k_atoms  <<<NUM_MOLS, 256, 0, stream>>>(z, W1, b1, W2, b2, ws, out);
  k_edges  <<<NUM_MOLS, 256, 0, stream>>>(z, bm, out);
}
